// GNNExtractor_67860483276910
// MI455X (gfx1250) — compile-verified, matrix operand order checked by reading, not run on hardware
//
#include <hip/hip_runtime.h>
#include <hip/hip_bf16.h>

typedef __attribute__((ext_vector_type(16))) _Float16 v16h;
typedef __attribute__((ext_vector_type(8)))  float    v8f;

#define DIM 64

// ---------------------------------------------------------------------------
// Edge scatter, 4-channel features: agg[dst] += x[src]  (agg pre-init with x)
// ---------------------------------------------------------------------------
__global__ void scatter4_kernel(const float4* __restrict__ x,
                                const int* __restrict__ src,
                                const int* __restrict__ dst,
                                float* __restrict__ agg, int nEdges) {
    int e = blockIdx.x * blockDim.x + threadIdx.x;
    if (e >= nEdges) return;
    int s = src[e], d = dst[e];
    float4 v = x[s];
    float* p = agg + (size_t)d * 4;
    atomicAdd(p + 0, v.x); atomicAdd(p + 1, v.y);
    atomicAdd(p + 2, v.z); atomicAdd(p + 3, v.w);
}

// ---------------------------------------------------------------------------
// Edge scatter, 64-channel features: 16 lanes per edge, float4 per lane.
// h table (25.6MB) is L2-resident -> gather + atomics run at L2 rates.
// ---------------------------------------------------------------------------
__global__ void scatter64_kernel(const float* __restrict__ h,
                                 const int* __restrict__ src,
                                 const int* __restrict__ dst,
                                 float* __restrict__ agg, int nEdges) {
    long t = (long)blockIdx.x * blockDim.x + threadIdx.x;
    int e = (int)(t >> 4);
    if (e >= nEdges) return;
    int c = ((int)t & 15) * 4;
    int s = src[e], d = dst[e];
    float4 v = *(const float4*)(h + (size_t)s * DIM + c);
    float* p = agg + (size_t)d * DIM + c;
    atomicAdd(p + 0, v.x); atomicAdd(p + 1, v.y);
    atomicAdd(p + 2, v.z); atomicAdd(p + 3, v.w);
}

// ---------------------------------------------------------------------------
// Y[n,d] = sum_f agg4[n,f] * w[d,f] + b[d]   (4 -> 64 lift, tiny K)
// ---------------------------------------------------------------------------
__global__ void lin4to64_kernel(const float* __restrict__ agg4,
                                const float* __restrict__ w,   // (64,4)
                                const float* __restrict__ b,   // (64)
                                float* __restrict__ y, int nNodes) {
    long i = (long)blockIdx.x * blockDim.x + threadIdx.x;
    if (i >= (long)nNodes * DIM) return;
    int n = (int)(i >> 6), d = (int)(i & 63);
    float4 a = *(const float4*)(agg4 + (size_t)n * 4);
    float4 ww = *(const float4*)(w + (size_t)d * 4);
    y[i] = fmaf(a.x, ww.x, fmaf(a.y, ww.y, fmaf(a.z, ww.z, fmaf(a.w, ww.w, b[d]))));
}

// ---------------------------------------------------------------------------
// WMMA node matmul: Y(n,64) = H(n,64) @ W(64,64)^T + bias
// block = 128 threads (4 waves), 64-node tile per block.
// Wave w owns out-columns [w*16, w*16+16): loads its B-fragments ONCE and
// reuses them across 4 node sub-tiles -> 8 v_wmma per wave per block.
// Fragment layout per CDNA5 ISA 7.12.2 (16-bit A, 32-bit C/D, wave32):
//   lane L: M/N = L&15, hi = L>>4; half e -> K = (e&7) + ((e>>3)<<4) + hi*8
//   C/D vgpr r: row M = r + hi*8, col N = L&15
// ---------------------------------------------------------------------------
__global__ void __launch_bounds__(128)
node_matmul64_wmma(const float* __restrict__ H, const float* __restrict__ W,
                   const float* __restrict__ bias, float* __restrict__ Y,
                   int nNodes) {
    __shared__ _Float16 sW[DIM * DIM];   // 8 KB
    __shared__ _Float16 sH[DIM * DIM];   // 8 KB (64-node tile)
    int tid = threadIdx.x;
    int wave = tid >> 5;                 // 0..3 -> output column tile
    int lane = tid & 31;
    int nodeBase = blockIdx.x * 64;
    bool full = (nodeBase + 64 <= nNodes);

    // stage W as f16 (float4 global loads)
    {
        const float4* W4 = (const float4*)W;
        for (int i = tid; i < DIM * DIM / 4; i += 128) {
            float4 v = W4[i];
            int o = i * 4;
            sW[o + 0] = (_Float16)v.x; sW[o + 1] = (_Float16)v.y;
            sW[o + 2] = (_Float16)v.z; sW[o + 3] = (_Float16)v.w;
        }
    }
    // stage 64-node H tile as f16
    if (full) {
        const float4* H4 = (const float4*)(H + (size_t)nodeBase * DIM);
        for (int i = tid; i < DIM * DIM / 4; i += 128) {
            float4 v = H4[i];
            int o = i * 4;
            sH[o + 0] = (_Float16)v.x; sH[o + 1] = (_Float16)v.y;
            sH[o + 2] = (_Float16)v.z; sH[o + 3] = (_Float16)v.w;
        }
    } else {
        for (int i = tid; i < DIM * DIM; i += 128) {
            int n = nodeBase + (i >> 6);
            sH[i] = (n < nNodes) ? (_Float16)H[(size_t)n * DIM + (i & 63)]
                                 : (_Float16)0.0f;
        }
    }
    __syncthreads();

    int m  = lane & 15;
    int hi = lane >> 4;
    int d  = wave * 16 + m;
    float bv = bias[d];

    // B fragments (K=0..31 and K=32..63), loaded once per wave
    v16h b0, b1;
#pragma unroll
    for (int e = 0; e < 16; ++e) {
        int k = hi * 8 + (e & 7) + ((e >> 3) << 4);
        b0[e] = sW[d * DIM + k];
        b1[e] = sW[d * DIM + 32 + k];
    }

#pragma unroll
    for (int st = 0; st < 4; ++st) {
        int rowBase = st * 16;
        v16h a0, a1;
#pragma unroll
        for (int e = 0; e < 16; ++e) {
            int k = hi * 8 + (e & 7) + ((e >> 3) << 4);
            a0[e] = sH[(rowBase + m) * DIM + k];
            a1[e] = sH[(rowBase + m) * DIM + 32 + k];
        }
        v8f acc = {};
        acc = __builtin_amdgcn_wmma_f32_16x16x32_f16(
            false, a0, false, b0, (short)0, acc, false, false);
        acc = __builtin_amdgcn_wmma_f32_16x16x32_f16(
            false, a1, false, b1, (short)0, acc, false, false);
        if (full) {
#pragma unroll
            for (int r = 0; r < 8; ++r) {
                int n = nodeBase + rowBase + r + hi * 8;
                Y[(size_t)n * DIM + d] = acc[r] + bv;
            }
        } else {
#pragma unroll
            for (int r = 0; r < 8; ++r) {
                int n = nodeBase + rowBase + r + hi * 8;
                if (n < nNodes) Y[(size_t)n * DIM + d] = acc[r] + bv;
            }
        }
    }
}

// ---------------------------------------------------------------------------
// Per-channel sum / sumsq over nodes, float4-vectorized.
// stride is a multiple of 64 -> each thread's 4-channel group is invariant:
// private accumulate, LDS ds_add flush, one global atomic per channel/block.
// ---------------------------------------------------------------------------
__global__ void stats64_kernel(const float* __restrict__ y, long total,
                               float* __restrict__ stats /*128, pre-zeroed*/) {
    __shared__ float s[DIM], ss[DIM];
    int tid = threadIdx.x;
    if (tid < DIM) { s[tid] = 0.f; ss[tid] = 0.f; }
    __syncthreads();
    long stride = (long)gridDim.x * blockDim.x * 4;
    long i0 = ((long)blockIdx.x * blockDim.x + tid) * 4;
    int c = (int)(i0 & 63);
    float ls0 = 0.f, ls1 = 0.f, ls2 = 0.f, ls3 = 0.f;
    float q0 = 0.f, q1 = 0.f, q2 = 0.f, q3 = 0.f;
    for (long i = i0; i < total; i += stride) {
        float4 v = *(const float4*)(y + i);
        ls0 += v.x; q0 += v.x * v.x;
        ls1 += v.y; q1 += v.y * v.y;
        ls2 += v.z; q2 += v.z * v.z;
        ls3 += v.w; q3 += v.w * v.w;
    }
    atomicAdd(&s[c + 0], ls0); atomicAdd(&ss[c + 0], q0);
    atomicAdd(&s[c + 1], ls1); atomicAdd(&ss[c + 1], q1);
    atomicAdd(&s[c + 2], ls2); atomicAdd(&ss[c + 2], q2);
    atomicAdd(&s[c + 3], ls3); atomicAdd(&ss[c + 3], q3);
    __syncthreads();
    if (tid < DIM) {
        atomicAdd(&stats[tid], s[tid]);
        atomicAdd(&stats[DIM + tid], ss[tid]);
    }
}

__global__ void bn_finalize_kernel(const float* __restrict__ stats,
                                   const float* __restrict__ g,
                                   const float* __restrict__ be,
                                   float* __restrict__ sshift, float invN) {
    int c = threadIdx.x;
    if (c >= DIM) return;
    float mean = stats[c] * invN;
    float var  = stats[DIM + c] * invN - mean * mean;
    float scale = g[c] * rsqrtf(var + 1e-5f);
    sshift[c] = scale;
    sshift[DIM + c] = be[c] - mean * scale;
}

// float4-vectorized BN + ReLU (total is a multiple of 64)
__global__ void bn_relu_apply_kernel(const float* __restrict__ src,
                                     const float* __restrict__ sshift,
                                     float* __restrict__ dst, long total) {
    long i = ((long)blockIdx.x * blockDim.x + threadIdx.x) * 4;
    if (i >= total) return;
    int c = (int)(i & 63);
    float4 v = *(const float4*)(src + i);
    v.x = fmaxf(fmaf(v.x, sshift[c + 0], sshift[DIM + c + 0]), 0.f);
    v.y = fmaxf(fmaf(v.y, sshift[c + 1], sshift[DIM + c + 1]), 0.f);
    v.z = fmaxf(fmaf(v.z, sshift[c + 2], sshift[DIM + c + 2]), 0.f);
    v.w = fmaxf(fmaf(v.w, sshift[c + 3], sshift[DIM + c + 3]), 0.f);
    *(float4*)(dst + i) = v;
}

// ---------------------------------------------------------------------------
// Pools (segment_sum by batch id)
// ---------------------------------------------------------------------------
__global__ void poolx_kernel(const float4* __restrict__ x,
                             const int* __restrict__ batch,
                             float* __restrict__ pool, int nNodes) {
    int n = blockIdx.x * blockDim.x + threadIdx.x;
    if (n >= nNodes) return;
    float4 v = x[n];
    float* p = pool + (size_t)batch[n] * 4;
    atomicAdd(p + 0, v.x); atomicAdd(p + 1, v.y);
    atomicAdd(p + 2, v.z); atomicAdd(p + 3, v.w);
}

__global__ void poolh_kernel(const float* __restrict__ h,
                             const int* __restrict__ batch,
                             float* __restrict__ pool, int nNodes) {
    long t = (long)blockIdx.x * blockDim.x + threadIdx.x;
    int n = (int)(t >> 4);
    if (n >= nNodes) return;
    int c = ((int)t & 15) * 4;
    float4 v = *(const float4*)(h + (size_t)n * DIM + c);
    float* p = pool + (size_t)batch[n] * DIM + c;
    atomicAdd(p + 0, v.x); atomicAdd(p + 1, v.y);
    atomicAdd(p + 2, v.z); atomicAdd(p + 3, v.w);
}

// ---------------------------------------------------------------------------
// Head: out = poolx@fc0^T + pool1@fc1^T + pool2@fc2^T (+biases)
// ---------------------------------------------------------------------------
__global__ void head1_kernel(const float* __restrict__ poolx,
                             const float* __restrict__ pool1,
                             const float* __restrict__ pool2,
                             const float* __restrict__ fc0w, const float* __restrict__ fc0b,
                             const float* __restrict__ fc1w, const float* __restrict__ fc1b,
                             const float* __restrict__ fc2w, const float* __restrict__ fc2b,
                             float* __restrict__ out, int nGraphs) {
    int i = blockIdx.x * blockDim.x + threadIdx.x;
    if (i >= nGraphs * DIM) return;
    int g = i >> 6, d = i & 63;
    float acc = fc0b[d] + fc1b[d] + fc2b[d];
#pragma unroll
    for (int f = 0; f < 4; ++f) acc = fmaf(poolx[g * 4 + f], fc0w[d * 4 + f], acc);
#pragma unroll 8
    for (int k = 0; k < DIM; ++k) acc = fmaf(pool1[g * DIM + k], fc1w[d * DIM + k], acc);
#pragma unroll 8
    for (int k = 0; k < DIM; ++k) acc = fmaf(pool2[g * DIM + k], fc2w[d * DIM + k], acc);
    out[i] = acc;
}

__global__ void head2_kernel(const float* __restrict__ out,
                             const float* __restrict__ piw, const float* __restrict__ pib,
                             const float* __restrict__ vfw, const float* __restrict__ vfb,
                             float* __restrict__ dout, int nGraphs) {
    int i = blockIdx.x * blockDim.x + threadIdx.x;
    if (i >= nGraphs * DIM) return;
    int g = i >> 6, d = i & 63;
    float ap = pib[d], av = vfb[d];
#pragma unroll 8
    for (int k = 0; k < DIM; ++k) {
        float o = out[g * DIM + k];
        ap = fmaf(o, piw[d * DIM + k], ap);
        av = fmaf(o, vfw[d * DIM + k], av);
    }
    dout[i] = fmaxf(ap, 0.f);
    dout[(size_t)nGraphs * DIM + i] = fmaxf(av, 0.f);
}

// ---------------------------------------------------------------------------
extern "C" void kernel_launch(void* const* d_in, const int* in_sizes, int n_in,
                              void* d_out, int out_size, void* d_ws, size_t ws_size,
                              hipStream_t stream) {
    const float* x      = (const float*)d_in[0];
    const int*   eidx   = (const int*)d_in[1];
    const int*   batch  = (const int*)d_in[2];
    const float* c1_w1  = (const float*)d_in[3];
    const float* c1_b1  = (const float*)d_in[4];
    const float* c1_g1  = (const float*)d_in[5];
    const float* c1_be1 = (const float*)d_in[6];
    const float* c1_w2  = (const float*)d_in[7];
    const float* c1_b2  = (const float*)d_in[8];
    const float* bn1_g  = (const float*)d_in[9];
    const float* bn1_b  = (const float*)d_in[10];
    const float* c2_w1  = (const float*)d_in[11];
    const float* c2_b1  = (const float*)d_in[12];
    const float* c2_g1  = (const float*)d_in[13];
    const float* c2_be1 = (const float*)d_in[14];
    const float* c2_w2  = (const float*)d_in[15];
    const float* c2_b2  = (const float*)d_in[16];
    const float* bn2_g  = (const float*)d_in[17];
    const float* bn2_b  = (const float*)d_in[18];
    const float* fc0_w  = (const float*)d_in[19];
    const float* fc0_b  = (const float*)d_in[20];
    const float* fc1_w  = (const float*)d_in[21];
    const float* fc1_b  = (const float*)d_in[22];
    const float* fc2_w  = (const float*)d_in[23];
    const float* fc2_b  = (const float*)d_in[24];
    const float* pi_w   = (const float*)d_in[25];
    const float* pi_b   = (const float*)d_in[26];
    const float* vf_w   = (const float*)d_in[27];
    const float* vf_b   = (const float*)d_in[28];

    const int nNodes  = in_sizes[0] / 4;          // 100000
    const int nEdges  = in_sizes[1] / 2;          // 3200000
    const int nGraphs = out_size / (2 * DIM);     // 512
    const int* src = eidx;
    const int* dst = eidx + nEdges;

    // workspace layout
    float* ws    = (float*)d_ws;
    float* aggX  = ws;                               // nNodes*4
    float* A     = aggX + (size_t)nNodes * 4;        // nNodes*64
    float* B     = A + (size_t)nNodes * DIM;         // nNodes*64
    float* C     = B + (size_t)nNodes * DIM;         // nNodes*64
    float* stats = C + (size_t)nNodes * DIM;         // 128
    float* sshift= stats + 2 * DIM;                  // 128
    float* poolx = sshift + 2 * DIM;                 // nGraphs*4
    float* pool1 = poolx + (size_t)nGraphs * 4;      // nGraphs*64
    float* pool2 = pool1 + (size_t)nGraphs * DIM;    // nGraphs*64
    float* outb  = pool2 + (size_t)nGraphs * DIM;    // nGraphs*64

    const long totalN64 = (long)nNodes * DIM;
    const float invN = 1.0f / (float)nNodes;
    const int B256 = 256;
    const int mmBlocks  = (nNodes + 63) / 64;
    const int nodeElB   = (int)((totalN64 + B256 - 1) / B256);
    const int vecElB    = (int)((totalN64 / 4 + B256 - 1) / B256);
    const int statBlocks = 512;

    // ================= conv1 =================
    hipMemcpyAsync(aggX, x, (size_t)nNodes * 4 * sizeof(float),
                   hipMemcpyDeviceToDevice, stream);
    scatter4_kernel<<<(nEdges + B256 - 1) / B256, B256, 0, stream>>>(
        (const float4*)x, src, dst, aggX, nEdges);
    lin4to64_kernel<<<nodeElB, B256, 0, stream>>>(aggX, c1_w1, c1_b1, A, nNodes);
    hipMemsetAsync(stats, 0, 2 * DIM * sizeof(float), stream);
    stats64_kernel<<<statBlocks, B256, 0, stream>>>(A, totalN64, stats);
    bn_finalize_kernel<<<1, DIM, 0, stream>>>(stats, c1_g1, c1_be1, sshift, invN);
    bn_relu_apply_kernel<<<vecElB, B256, 0, stream>>>(A, sshift, A, totalN64);
    node_matmul64_wmma<<<mmBlocks, 128, 0, stream>>>(A, c1_w2, c1_b2, C, nNodes);
    hipMemsetAsync(stats, 0, 2 * DIM * sizeof(float), stream);
    stats64_kernel<<<statBlocks, B256, 0, stream>>>(C, totalN64, stats);
    bn_finalize_kernel<<<1, DIM, 0, stream>>>(stats, bn1_g, bn1_b, sshift, invN);
    bn_relu_apply_kernel<<<vecElB, B256, 0, stream>>>(C, sshift, B, totalN64);

    // ================= conv2 =================
    hipMemcpyAsync(A, B, (size_t)totalN64 * sizeof(float),
                   hipMemcpyDeviceToDevice, stream);
    {
        long t = (long)nEdges * 16;
        scatter64_kernel<<<(int)((t + B256 - 1) / B256), B256, 0, stream>>>(
            B, src, dst, A, nEdges);
    }
    node_matmul64_wmma<<<mmBlocks, 128, 0, stream>>>(A, c2_w1, c2_b1, C, nNodes);
    hipMemsetAsync(stats, 0, 2 * DIM * sizeof(float), stream);
    stats64_kernel<<<statBlocks, B256, 0, stream>>>(C, totalN64, stats);
    bn_finalize_kernel<<<1, DIM, 0, stream>>>(stats, c2_g1, c2_be1, sshift, invN);
    bn_relu_apply_kernel<<<vecElB, B256, 0, stream>>>(C, sshift, C, totalN64);
    node_matmul64_wmma<<<mmBlocks, 128, 0, stream>>>(C, c2_w2, c2_b2, A, nNodes);
    hipMemsetAsync(stats, 0, 2 * DIM * sizeof(float), stream);
    stats64_kernel<<<statBlocks, B256, 0, stream>>>(A, totalN64, stats);
    bn_finalize_kernel<<<1, DIM, 0, stream>>>(stats, bn2_g, bn2_b, sshift, invN);
    bn_relu_apply_kernel<<<vecElB, B256, 0, stream>>>(A, sshift, A, totalN64);

    // ================= pooling + head =================
    hipMemsetAsync(poolx, 0,
                   ((size_t)nGraphs * 4 + 2 * (size_t)nGraphs * DIM) * sizeof(float),
                   stream);
    poolx_kernel<<<(nNodes + B256 - 1) / B256, B256, 0, stream>>>(
        (const float4*)x, batch, poolx, nNodes);
    {
        long t = (long)nNodes * 16;
        int blk = (int)((t + B256 - 1) / B256);
        poolh_kernel<<<blk, B256, 0, stream>>>(B, batch, pool1, nNodes); // h1
        poolh_kernel<<<blk, B256, 0, stream>>>(A, batch, pool2, nNodes); // h2
    }
    int headBlk = (nGraphs * DIM + B256 - 1) / B256;
    head1_kernel<<<headBlk, B256, 0, stream>>>(poolx, pool1, pool2,
                                               fc0_w, fc0_b, fc1_w, fc1_b,
                                               fc2_w, fc2_b, outb, nGraphs);
    head2_kernel<<<headBlk, B256, 0, stream>>>(outb, pi_w, pi_b, vf_w, vf_b,
                                               (float*)d_out, nGraphs);
}
